// EDFFN_50706383896924
// MI455X (gfx1250) — matrix operand order repaired and checked by reading, not compiled, not run on hardware
//
#include <hip/hip_runtime.h>

// ---------------- problem constants ----------------
#define Bn    4
#define Cn    48
#define Hn    256
#define Wn    256
#define HIDn  192
#define HID2n 384
#define TH    8      // tile height  (== FFT patch)
#define TW    16     // tile width   (two FFT patches)
#define NPIX  180    // 10x18 halo pixels
#define NPIXP 192    // padded to 12 groups of 16
#define KPAD  64     // K=48 padded to 64 for bf16 WMMA
#define YSTR  392    // Y row stride in bf16 elems (384 + 8 pad: 196 dwords == 4 mod 64 banks)
#define NTHREADS 512 // 16 wave32

typedef __attribute__((ext_vector_type(16))) __bf16 v16bf;
typedef __attribute__((ext_vector_type(8)))  float  v8f;

union V16BF { v16bf v; __bf16 e[16]; };
union V8F   { v8f   v; float  e[8];  };
union PK8   { __bf16 h[8]; int4 q; };

// ---------------- LDS carving (bytes) ----------------
//   [0 .. 49152)        Win [384][64] bf16 (st1)    ->  H [128][192] bf16 (st2-3)
//   [49152 .. 199680)   Y [192 px][392 ch] bf16 (st1-2)
//                         -> O [48][128] f32 @49152 (st3-4), Spec [96][8][5][2] f32 @73728
//   [199680 .. 224256)  XN [192][64] bf16 (st1)
//   [224256 .. )        persistent: Wout, wdw, gains, biases, ln params
#define OFF_WIN   0
#define OFF_H     0
#define OFF_Y     49152
#define OFF_O     49152
#define OFF_SPEC  73728
#define OFF_XN    199680
#define OFF_WOUT  224256
#define OFF_WDW   242688
#define OFF_GAIN  256512
#define OFF_BIN   264192
#define OFF_BDW   265728
#define OFF_BOUT  267264
#define OFF_LNW   267520
#define OFF_LNB   267776
#define LDS_BYTES 268032

__device__ __forceinline__ float softplusf(float x) {
  if (x > 20.f) return x;
  if (x < -20.f) return expf(x);
  return log1pf(expf(x));
}

// DFT-8 twiddles: cos/sin(2*pi*k/8)
__constant__ float CT8[8] = {1.f, 0.70710678f, 0.f, -0.70710678f, -1.f, -0.70710678f, 0.f, 0.70710678f};
__constant__ float ST8[8] = {0.f, 0.70710678f, 1.f, 0.70710678f, 0.f, -0.70710678f, -1.f, -0.70710678f};

// Load A fragment (16-bit A 16x32 lane layout): row = arow, chunks [+0..7] and [+16..23]
__device__ __forceinline__ void load_afrag(V16BF& a, const __bf16* arow) {
#pragma unroll
  for (int e = 0; e < 8; ++e) { a.e[e] = arow[e]; a.e[e + 8] = arow[e + 16]; }
}
// Load B fragment (16-bit B 32x16 lane layout): col = bcol, chunks [+0..7] and [+16..23]
__device__ __forceinline__ void load_bfrag(V16BF& bm, const __bf16* bcol) {
#pragma unroll
  for (int e = 0; e < 8; ++e) { bm.e[e] = bcol[e]; bm.e[e + 8] = bcol[e + 16]; }
}

extern "C" __global__ __launch_bounds__(NTHREADS, 1)
void EDFFN_fused_kernel(const float* __restrict__ x,
                        const float* __restrict__ ln_w, const float* __restrict__ ln_b,
                        const float* __restrict__ w_in, const float* __restrict__ b_in,
                        const float* __restrict__ w_dw, const float* __restrict__ b_dw,
                        const float* __restrict__ w_out, const float* __restrict__ b_out,
                        const float* __restrict__ gain,
                        float* __restrict__ out) {
  extern __shared__ char smem[];
  __bf16* winl  = (__bf16*)(smem + OFF_WIN);   // [384][64]
  __bf16* hbuf  = (__bf16*)(smem + OFF_H);     // [128 px][192 ch]
  __bf16* ybuf  = (__bf16*)(smem + OFF_Y);     // [192 px][392]
  float*  obuf  = (float*) (smem + OFF_O);     // [48][128]
  float*  spec  = (float*) (smem + OFF_SPEC);  // [96][8][5][2]
  __bf16* xnl   = (__bf16*)(smem + OFF_XN);    // [192 px][64 ch]
  __bf16* woutl = (__bf16*)(smem + OFF_WOUT);  // [48][192]
  float*  wdwl  = (float*) (smem + OFF_WDW);   // [384][9]
  float*  gl    = (float*) (smem + OFF_GAIN);  // [48][8][5] softplus(gain)
  float*  binl  = (float*) (smem + OFF_BIN);   // [384]
  float*  bdwl  = (float*) (smem + OFF_BDW);   // [384]
  float*  boutl = (float*) (smem + OFF_BOUT);  // [48]
  float*  lnwl  = (float*) (smem + OFF_LNW);   // [48]
  float*  lnbl  = (float*) (smem + OFF_LNB);   // [48]

  const int tid  = threadIdx.x;
  const int wave = tid >> 5;
  const int lane = tid & 31;
  const int mlo  = lane & 15;       // WMMA: B column / A row (mod 16)
  const int hi   = lane >> 4;       // WMMA: lane half

  const int wg = blockIdx.x;
  const int tx = wg & 15;           // W/TW = 16 tiles
  const int ty = (wg >> 4) & 31;    // H/TH = 32 tiles
  const int b  = wg >> 9;           // batch

  const int gy_base = ty * TH;
  const int gx_base = tx * TW;

  // ---------------- stage 0: preload weights into LDS ----------------
  for (int i = tid; i < HID2n * KPAD; i += NTHREADS) {
    int o = i >> 6, c = i & 63;
    winl[i] = (c < Cn) ? (__bf16)w_in[o * Cn + c] : (__bf16)0.f;
  }
  for (int i = tid; i < Cn * HIDn; i += NTHREADS) woutl[i] = (__bf16)w_out[i];
  for (int i = tid; i < HID2n * 9; i += NTHREADS) wdwl[i] = w_dw[i];
  for (int i = tid; i < Cn * 40;   i += NTHREADS) gl[i]   = softplusf(gain[i]);
  for (int i = tid; i < HID2n;     i += NTHREADS) { binl[i] = b_in[i]; bdwl[i] = b_dw[i]; }
  for (int i = tid; i < Cn;        i += NTHREADS) { boutl[i] = b_out[i]; lnwl[i] = ln_w[i]; lnbl[i] = ln_b[i]; }
  __syncthreads();

  // ---------------- stage 1a: LayerNorm over channels for halo pixels ----------------
  for (int p = tid; p < NPIXP; p += NTHREADS) {
    bool live = false;
    float v[Cn];
    int gy = 0, gx = 0;
    if (p < NPIX) {
      int hr = p / 18, hc = p - hr * 18;
      gy = gy_base + hr - 1;
      gx = gx_base + hc - 1;
      live = (gy >= 0) && (gy < Hn) && (gx >= 0) && (gx < Wn);
    }
    __bf16* xrow = xnl + p * KPAD;
    if (live) {
      const float* xp = x + ((size_t)b * Cn * Hn + gy) * Wn + gx;
      float s = 0.f;
#pragma unroll
      for (int c = 0; c < Cn; ++c) { v[c] = xp[(size_t)c * Hn * Wn]; s += v[c]; }
      float mu = s * (1.f / Cn);
      float vs = 0.f;
#pragma unroll
      for (int c = 0; c < Cn; ++c) { float d = v[c] - mu; vs += d * d; }
      float rstd = rsqrtf(vs * (1.f / Cn) + 1e-5f);
#pragma unroll
      for (int c0 = 0; c0 < KPAD; c0 += 8) {
        PK8 pk;
#pragma unroll
        for (int c = 0; c < 8; ++c) {
          int cc = c0 + c;
          pk.h[c] = (cc < Cn) ? (__bf16)(((v[cc] - mu) * rstd) * lnwl[cc] + lnbl[cc])
                              : (__bf16)0.f;
        }
        *(int4*)(xrow + c0) = pk.q;
      }
    } else {
      int4 z = make_int4(0, 0, 0, 0);
#pragma unroll
      for (int c0 = 0; c0 < KPAD; c0 += 8) *(int4*)(xrow + c0) = z;
    }
  }
  __syncthreads();

  // ---------------- stage 1b: proj_in GEMM via WMMA bf16 ----------------
  // Y[px][ch] (+bias) = (Win[384 x 64] * XN[64 x 192])^T-ish store.
  // 144 pair-tasks: (mtp, ng) computes M-tiles mtp and mtp+12 sharing the B fragment.
  // Two independent accumulator chains fill the WMMA RAW hazard slots.
  for (int pt = wave; pt < 12 * 12; pt += 16) {
    int mtp = pt / 12, ng = pt - mtp * 12;
    int o0a = mtp * 16, o0b = (mtp + 12) * 16, p0 = ng * 16;
    V8F acc0, acc1;
#pragma unroll
    for (int r = 0; r < 8; ++r) {
      acc0.e[r] = binl[o0a + r + 8 * hi];
      acc1.e[r] = binl[o0b + r + 8 * hi];
    }
#pragma unroll
    for (int ks = 0; ks < 2; ++ks) {
      int k0 = ks * 32;
      V16BF a0, a1, bm;
      load_bfrag(bm, xnl + (p0 + mlo) * KPAD + k0 + hi * 8);
      load_afrag(a0, winl + (o0a + mlo) * KPAD + k0 + hi * 8);
      load_afrag(a1, winl + (o0b + mlo) * KPAD + k0 + hi * 8);
      acc0.v = __builtin_amdgcn_wmma_f32_16x16x32_bf16(false, a0.v, false, bm.v,
                                                       (short)0, acc0.v, false, false);
      acc1.v = __builtin_amdgcn_wmma_f32_16x16x32_bf16(false, a1.v, false, bm.v,
                                                       (short)0, acc1.v, false, false);
    }
    // Packed stores: lane's 8 output channels are contiguous in Y[px][ch].
    __bf16* yrow = ybuf + (p0 + mlo) * YSTR + 8 * hi;
    PK8 pk0, pk1;
#pragma unroll
    for (int r = 0; r < 8; ++r) { pk0.h[r] = (__bf16)acc0.e[r]; pk1.h[r] = (__bf16)acc1.e[r]; }
    *(int4*)(yrow + o0a) = pk0.q;
    *(int4*)(yrow + o0b) = pk1.q;
  }
  __syncthreads();

  // ---------------- stage 2: depthwise 3x3 + bias + gated exact GELU ----------------
  // ch is the lane-fast index: Y reads and H writes are lane-contiguous.
  for (int i = tid; i < HIDn * TH * TW; i += NTHREADS) {
    int pix = i / HIDn, ch = i - pix * HIDn;
    int r = pix >> 4, c = pix & 15;
    int pcen = (r + 1) * 18 + (c + 1);
    int gy0 = gy_base + r, gx0 = gx_base + c;
    float z1 = bdwl[ch], z2 = bdwl[ch + HIDn];
    const float* wd1 = wdwl + ch * 9;
    const float* wd2 = wdwl + (ch + HIDn) * 9;
#pragma unroll
    for (int dr = -1; dr <= 1; ++dr) {
#pragma unroll
      for (int dc = -1; dc <= 1; ++dc) {
        int gy = gy0 + dr, gx = gx0 + dc;
        if (gy >= 0 && gy < Hn && gx >= 0 && gx < Wn) {
          const __bf16* yp = ybuf + (pcen + dr * 18 + dc) * YSTR + ch;
          int wi = (dr + 1) * 3 + (dc + 1);
          z1 += wd1[wi] * (float)yp[0];
          z2 += wd2[wi] * (float)yp[HIDn];
        }
      }
    }
    float g1 = 0.5f * z1 * (1.f + erff(z1 * 0.70710678f));
    hbuf[pix * HIDn + ch] = (__bf16)(g1 * z2);
  }
  __syncthreads();

  // ---------------- stage 3: proj_out GEMM via WMMA bf16 ----------------
  // O[48 x 128] = Wout[48 x 192] * H[192 x 128] + b_out
  // 12 pair-tasks: (mt, ngp) computes N-groups ngp and ngp+4 sharing the A fragment.
  for (int pt = wave; pt < 12; pt += 16) {
    int mt = pt >> 2, ngp = pt & 3;
    int o0 = mt * 16, p0a = ngp * 16, p0b = (ngp + 4) * 16;
    V8F acc0, acc1;
#pragma unroll
    for (int r = 0; r < 8; ++r) { acc0.e[r] = boutl[o0 + r + 8 * hi]; acc1.e[r] = acc0.e[r]; }
#pragma unroll
    for (int ks = 0; ks < 6; ++ks) {
      int k0 = ks * 32;
      V16BF a, b0, b1;
      load_afrag(a, woutl + (o0 + mlo) * HIDn + k0 + hi * 8);
      load_bfrag(b0, hbuf + (p0a + mlo) * HIDn + k0 + hi * 8);
      load_bfrag(b1, hbuf + (p0b + mlo) * HIDn + k0 + hi * 8);
      acc0.v = __builtin_amdgcn_wmma_f32_16x16x32_bf16(false, a.v, false, b0.v,
                                                       (short)0, acc0.v, false, false);
      acc1.v = __builtin_amdgcn_wmma_f32_16x16x32_bf16(false, a.v, false, b1.v,
                                                       (short)0, acc1.v, false, false);
    }
#pragma unroll
    for (int r = 0; r < 8; ++r) {
      obuf[(o0 + r + 8 * hi) * 128 + p0a + mlo] = acc0.e[r];
      obuf[(o0 + r + 8 * hi) * 128 + p0b + mlo] = acc1.e[r];
    }
  }
  __syncthreads();

  // ---------------- stage 4: patchwise rfft2 * softplus(gain) * irfft2 + residual ----
  // Phase A: row rFFT8. task = (cp = ch*2+patch, row); 96*8 = 768 tasks.
  for (int t = tid; t < 96 * 8; t += NTHREADS) {
    int cp = t >> 3, row = t & 7;
    int ch = cp >> 1, patch = cp & 1;
    const float* f = obuf + ch * 128 + row * 16 + patch * 8;
    float* sp = spec + ((cp * 8 + row) * 5) * 2;
#pragma unroll
    for (int v = 0; v < 5; ++v) {
      float re = 0.f, im = 0.f;
#pragma unroll
      for (int j = 0; j < 8; ++j) {
        int tw = (j * v) & 7;
        re += f[j] * CT8[tw];
        im -= f[j] * ST8[tw];
      }
      sp[v * 2 + 0] = re;
      sp[v * 2 + 1] = im;
    }
  }
  __syncthreads();

  // Phase B: per (cp, v): column DFT8 -> gain -> inverse DFT8 (x 1/8). 480 tasks.
  if (tid < 96 * 5) {
    int cp = tid / 5, v = tid - cp * 5;
    int ch = cp >> 1;
    float cr[8], ci[8], Fr[8], Fi[8];
#pragma unroll
    for (int u = 0; u < 8; ++u) {
      cr[u] = spec[((cp * 8 + u) * 5 + v) * 2 + 0];
      ci[u] = spec[((cp * 8 + u) * 5 + v) * 2 + 1];
    }
#pragma unroll
    for (int k = 0; k < 8; ++k) {
      float fr = 0.f, fi = 0.f;
#pragma unroll
      for (int u = 0; u < 8; ++u) {
        int tw = (u * k) & 7;
        fr += cr[u] * CT8[tw] + ci[u] * ST8[tw];
        fi += ci[u] * CT8[tw] - cr[u] * ST8[tw];
      }
      float gm = gl[ch * 40 + k * 5 + v];
      Fr[k] = fr * gm;
      Fi[k] = fi * gm;
    }
#pragma unroll
    for (int u = 0; u < 8; ++u) {
      float rr = 0.f, ii = 0.f;
#pragma unroll
      for (int k = 0; k < 8; ++k) {
        int tw = (u * k) & 7;
        rr += Fr[k] * CT8[tw] - Fi[k] * ST8[tw];
        ii += Fr[k] * ST8[tw] + Fi[k] * CT8[tw];
      }
      spec[((cp * 8 + u) * 5 + v) * 2 + 0] = rr * 0.125f;
      spec[((cp * 8 + u) * 5 + v) * 2 + 1] = ii * 0.125f;
    }
  }
  __syncthreads();

  // Phase C: row irFFT8 (x 1/8) + residual add, write out. 768 tasks.
  for (int t = tid; t < 96 * 8; t += NTHREADS) {
    int cp = t >> 3, row = t & 7;
    int ch = cp >> 1, patch = cp & 1;
    const float* sp = spec + ((cp * 8 + row) * 5) * 2;
    float x0 = sp[0];
    float x4 = sp[8];
    int gy = gy_base + row;
    int gx0 = gx_base + patch * 8;
    const float* xr   = x   + ((size_t)(b * Cn + ch) * Hn + gy) * Wn + gx0;
    float*       orow = out + ((size_t)(b * Cn + ch) * Hn + gy) * Wn + gx0;
#pragma unroll
    for (int j = 0; j < 8; ++j) {
      float s = x0 + ((j & 1) ? -x4 : x4);
#pragma unroll
      for (int v = 1; v < 4; ++v) {
        int tw = (j * v) & 7;
        s += 2.f * (sp[v * 2 + 0] * CT8[tw] - sp[v * 2 + 1] * ST8[tw]);
      }
      orow[j] = xr[j] + s * 0.125f;
    }
  }
}

extern "C" void kernel_launch(void* const* d_in, const int* in_sizes, int n_in,
                              void* d_out, int out_size, void* d_ws, size_t ws_size,
                              hipStream_t stream) {
  (void)in_sizes; (void)n_in; (void)d_ws; (void)ws_size; (void)out_size;
  const float* x     = (const float*)d_in[0];
  const float* ln_w  = (const float*)d_in[1];
  const float* ln_b  = (const float*)d_in[2];
  const float* w_in  = (const float*)d_in[3];
  const float* b_in  = (const float*)d_in[4];
  const float* w_dw  = (const float*)d_in[5];
  const float* b_dw  = (const float*)d_in[6];
  const float* w_out = (const float*)d_in[7];
  const float* b_out = (const float*)d_in[8];
  const float* gain  = (const float*)d_in[9];
  float* out = (float*)d_out;

  (void)hipFuncSetAttribute((const void*)EDFFN_fused_kernel,
                            hipFuncAttributeMaxDynamicSharedMemorySize, LDS_BYTES);

  dim3 grid(Bn * (Hn / TH) * (Wn / TW));  // 4 * 32 * 16 = 2048
  dim3 block(NTHREADS);
  EDFFN_fused_kernel<<<grid, block, LDS_BYTES, stream>>>(
      x, ln_w, ln_b, w_in, b_in, w_dw, b_dw, w_out, b_out, gain, out);
}
